// GXN_72602127172008
// MI455X (gfx1250) — matrix operand order repaired and checked by reading, not compiled, not run on hardware
//
#include <hip/hip_runtime.h>
#include <hip/hip_bf16.h>
#include <math.h>

// ---------------- problem constants (match reference) ----------------
#define BGRAPH   64
#define NNODE    1500
#define NT       (BGRAPH * NNODE)      // 96000
#define F_IN     128
#define KPOOL    30
#define TOTLAT   97
#define OUTD     128

typedef __attribute__((ext_vector_type(16))) __bf16 v16bf;
typedef __attribute__((ext_vector_type(8)))  float  v8f;

union BfPack {
    unsigned int u[8];
    v16bf        v;
};

static __device__ __forceinline__ unsigned int f2bf(float f) {
    unsigned int u = __float_as_uint(f);
    // round-to-nearest-even bf16
    u = (u + 0x7FFFu + ((u >> 16) & 1u)) >> 16;
    return u & 0xFFFFu;
}
static __device__ __forceinline__ unsigned int packbf(float lo, float hi) {
    return f2bf(lo) | (f2bf(hi) << 16);
}

// ---------------- degree kernels ----------------
__global__ void init_deg_kernel(float* deg) {
    int i = blockIdx.x * blockDim.x + threadIdx.x;
    if (i < NT) deg[i] = 1.0f;                      // +1 self loop
}

__global__ void deg_scatter_kernel(const int* edge_dst, float* deg, int E) {
    int i = blockIdx.x * blockDim.x + threadIdx.x;
    if (i < E) atomicAdd(&deg[edge_dst[i]], 1.0f);
}

// ---------------- agg = h + scatter_add(h[src] -> dst) ----------------
// Layer 1 fast path: stride==K==128, off==0, 16B-aligned -> float4 moves.
__global__ void agg_init_vec4_kernel(float4* agg, const float4* h) {
    int i = blockIdx.x * blockDim.x + threadIdx.x;   // NT * 32 float4s
    if (i < NT * (F_IN / 4)) agg[i] = h[i];
}

__global__ void agg_scatter_vec4_kernel(float* __restrict__ agg,
                                        const float* __restrict__ h,
                                        const int* __restrict__ src,
                                        const int* __restrict__ dst, int E) {
    long long i = (long long)blockIdx.x * blockDim.x + threadIdx.x;
    long long total = (long long)E * (F_IN / 4);
    if (i < total) {
        int e = (int)(i >> 5);                       // / 32 chunks
        int c = (int)(i & 31);
        const float4 v = ((const float4*)(h + (long long)src[e] * F_IN))[c];
        float* d = agg + (long long)dst[e] * F_IN + c * 4;
        atomicAdd(d + 0, v.x);
        atomicAdd(d + 1, v.y);
        atomicAdd(d + 2, v.z);
        atomicAdd(d + 3, v.w);
    }
}

// Generic path (layers 2-4 read a column slice of feat: stride TOTLAT, off).
__global__ void agg_init_kernel(float* agg, const float* h, int stride, int off, int K) {
    long long i = (long long)blockIdx.x * blockDim.x + threadIdx.x;
    long long total = (long long)NT * K;
    if (i < total) {
        int row = (int)(i / K), f = (int)(i % K);
        agg[i] = h[(long long)row * stride + off + f];
    }
}

__global__ void agg_scatter_kernel(float* agg, const float* h,
                                   const int* src, const int* dst,
                                   int stride, int off, int K, int E) {
    long long i = (long long)blockIdx.x * blockDim.x + threadIdx.x;
    long long total = (long long)E * K;
    if (i < total) {
        int e = (int)(i / K), f = (int)(i % K);
        float v = h[(long long)src[e] * stride + off + f];
        atomicAdd(&agg[(long long)dst[e] * K + f], v);
    }
}

// ---------------- WMMA GEMM + tanh(x/deg) epilogue ----------------
// D[96000 x NOUT] = tanh( (agg[96000 x K] @ W[K x NOUT] + b) / deg ),
// written into feat[:, colOff : colOff+NOUT] (row stride TOTLAT).
// Compile-time (K, NOUT): fully unrolled, no branches around v_wmma.
// W staged in LDS as bf16 pairs pre-arranged in the B-operand layout:
//   sWp[t*32 + n] packs rows {2t, 2t+1} (bf16 lo/hi) of column n, zero-padded.
template <int K, int NOUT>
__global__ __launch_bounds__(256)
void gemm_tanh_wmma_kernel(const float* __restrict__ agg,
                           const float* __restrict__ W,
                           const float* __restrict__ bias,
                           const float* __restrict__ deg,
                           float* __restrict__ feat,
                           int colOff) {
    constexpr int NTILES = (NOUT + 15) / 16;
    constexpr int KSTEPS = K / 32;
    __shared__ unsigned int sWp[(K / 2) * 32];

    const int tid  = threadIdx.x;
    const int lane = tid & 31;
    const int wave = tid >> 5;

    // cooperative LDS fill (bf16-pack + zero-pad columns >= NOUT)
    for (int i = tid; i < (K / 2) * 32; i += 256) {
        int t = i >> 5, n = i & 31;
        float lo = (n < NOUT) ? W[(2 * t)     * NOUT + n] : 0.0f;
        float hi = (n < NOUT) ? W[(2 * t + 1) * NOUT + n] : 0.0f;
        sWp[i] = packbf(lo, hi);
    }
    __syncthreads();

    const int mtile = blockIdx.x * 8 + wave;        // 750 blocks * 8 waves = 6000 tiles
    const int m     = lane & 15;                    // row within tile
    const int g     = lane >> 4;                    // lane-group (K-half select)
    const int row   = mtile * 16 + m;
    const int nn    = lane & 15;                    // column within N tile

    v8f acc[NTILES];
#pragma unroll
    for (int nt = 0; nt < NTILES; ++nt)
        acc[nt] = (v8f){0.f, 0.f, 0.f, 0.f, 0.f, 0.f, 0.f, 0.f};

#pragma unroll
    for (int kb = 0; kb < KSTEPS; ++kb) {
        // ---- A tile: 16x32 bf16, ISA layout:
        // lanes 0-15 (g=0): VGPR 0-3 -> K 0..7,  VGPR 4-7 -> K 16..23
        // lanes 16-31(g=1): VGPR 0-3 -> K 8..15, VGPR 4-7 -> K 24..31
        // Each half is 8 consecutive floats -> two aligned float4 loads.
        BfPack a;
        const float* prow = agg + (long long)row * K + (kb << 5) + (g << 3);
        float4 q0 = ((const float4*)prow)[0];
        float4 q1 = ((const float4*)prow)[1];
        float4 q2 = ((const float4*)(prow + 16))[0];
        float4 q3 = ((const float4*)(prow + 16))[1];
        a.u[0] = packbf(q0.x, q0.y); a.u[1] = packbf(q0.z, q0.w);
        a.u[2] = packbf(q1.x, q1.y); a.u[3] = packbf(q1.z, q1.w);
        a.u[4] = packbf(q2.x, q2.y); a.u[5] = packbf(q2.z, q2.w);
        a.u[6] = packbf(q3.x, q3.y); a.u[7] = packbf(q3.z, q3.w);

#pragma unroll
        for (int nt = 0; nt < NTILES; ++nt) {
            // ---- B tile: 32x16 bf16 from pre-packed LDS:
            // VGPR r packs rows {kb*32 + g*8 + 2r (+8 if r>=4), +1}, lane selects column
            BfPack b;
#pragma unroll
            for (int r = 0; r < 8; ++r) {
                int t = (kb << 4) + (g << 2) + r + ((r >> 2) << 2);
                b.u[r] = sWp[t * 32 + (nt << 4) + nn];
            }
            acc[nt] = __builtin_amdgcn_wmma_f32_16x16x32_bf16(
                false, a.v, false, b.v, (short)0, acc[nt], false, false);
        }
    }

    // epilogue: C layout -> VGPR v, lanes 0-15: M=v, lanes 16-31: M=v+8; N=lane&15
#pragma unroll
    for (int nt = 0; nt < NTILES; ++nt) {
        const int gn = (nt << 4) + nn;
#pragma unroll
        for (int v = 0; v < 8; ++v) {
            int gm = mtile * 16 + v + ((lane >> 4) << 3);
            if (gn < NOUT) {
                float x = (acc[nt][v] + bias[gn]) / deg[gm];
                feat[(long long)gm * TOTLAT + colOff + gn] = tanhf(x);
            }
        }
    }
}

// ---------------- sortpooling: top-30 per graph by feat[:,96] ----------------
__global__ void sortpool_kernel(const float* __restrict__ feat, int* __restrict__ gidx) {
    __shared__ float vals[NNODE];
    __shared__ float sval[256];
    __shared__ int   sidx[256];

    const int b = blockIdx.x, tid = threadIdx.x;
    for (int i = tid; i < NNODE; i += 256)
        vals[i] = feat[(long long)(b * NNODE + i) * TOTLAT + (TOTLAT - 1)];
    __syncthreads();

    for (int j = 0; j < KPOOL; ++j) {
        float best = -INFINITY;
        int   bi   = 1 << 30;
        for (int i = tid; i < NNODE; i += 256) {
            float v = vals[i];
            if (v > best || (v == best && i < bi)) { best = v; bi = i; }
        }
        sval[tid] = best; sidx[tid] = bi;
        __syncthreads();
        for (int s = 128; s > 0; s >>= 1) {
            if (tid < s) {
                if (sval[tid + s] > sval[tid] ||
                    (sval[tid + s] == sval[tid] && sidx[tid + s] < sidx[tid])) {
                    sval[tid] = sval[tid + s]; sidx[tid] = sidx[tid + s];
                }
            }
            __syncthreads();
        }
        if (tid == 0) {
            gidx[b * KPOOL + j] = b * NNODE + sidx[0];
            vals[sidx[0]] = -INFINITY;
        }
        __syncthreads();
    }
}

// ---------------- conv1: per-slot linear [B,K,97] x [16,97] -> relu [B,16,K] ----
__global__ void conv1_kernel(const float* __restrict__ feat, const int* __restrict__ gidx,
                             const float* __restrict__ w, const float* __restrict__ bb,
                             float* __restrict__ c1) {
    int idx = blockIdx.x * blockDim.x + threadIdx.x;
    if (idx >= BGRAPH * 16 * KPOOL) return;
    int b = idx / (16 * KPOOL);
    int r = idx % (16 * KPOOL);
    int o = r / KPOOL, k = r % KPOOL;
    long long row = gidx[b * KPOOL + k];
    float s = bb[o];
    for (int d = 0; d < TOTLAT; ++d)
        s += feat[row * TOTLAT + d] * w[o * TOTLAT + d];
    c1[idx] = fmaxf(s, 0.0f);                        // layout [b][o][k]
}

// ---------- maxpool(2,2) fused into conv2: [B,16,15] -> conv1d(k=5) -> relu [B,32,11]
__global__ void conv2_kernel(const float* __restrict__ c1,
                             const float* __restrict__ w, const float* __restrict__ bb,
                             float* __restrict__ c2) {
    int idx = blockIdx.x * blockDim.x + threadIdx.x;
    if (idx >= BGRAPH * 32 * 11) return;
    int b = idx / (32 * 11);
    int r = idx % (32 * 11);
    int oc = r / 11, t = r % 11;
    float s = bb[oc];
    for (int ic = 0; ic < 16; ++ic) {
        const float* base = c1 + (b * 16 + ic) * KPOOL;
        for (int j = 0; j < 5; ++j) {
            int p = t + j;
            float m = fmaxf(base[2 * p], base[2 * p + 1]);
            s += m * w[oc * 80 + ic * 5 + j];
        }
    }
    c2[(b * 32 + oc) * 11 + t] = fmaxf(s, 0.0f);
}

// ---------------- dense: [B,352] @ [352,128] + b -> relu ----------------
__global__ void dense_kernel(const float* __restrict__ c2,
                             const float* __restrict__ w, const float* __restrict__ bb,
                             float* __restrict__ out) {
    int idx = blockIdx.x * blockDim.x + threadIdx.x;
    if (idx >= BGRAPH * OUTD) return;
    int b = idx / OUTD, o = idx % OUTD;
    float s = bb[o];
    for (int i = 0; i < 352; ++i)
        s += c2[b * 352 + i] * w[i * OUTD + o];
    out[idx] = fmaxf(s, 0.0f);
}

// ---------------- host-side launcher ----------------
extern "C" void kernel_launch(void* const* d_in, const int* in_sizes, int n_in,
                              void* d_out, int out_size, void* d_ws, size_t ws_size,
                              hipStream_t stream) {
    const float* node_feat = (const float*)d_in[0];
    const int*   edge_src  = (const int*)d_in[1];
    const int*   edge_dst  = (const int*)d_in[2];
    const float* Wl[4] = {(const float*)d_in[3], (const float*)d_in[5],
                          (const float*)d_in[7], (const float*)d_in[9]};
    const float* bl[4] = {(const float*)d_in[4], (const float*)d_in[6],
                          (const float*)d_in[8], (const float*)d_in[10]};
    const float* conv1_w = (const float*)d_in[11];
    const float* conv1_b = (const float*)d_in[12];
    const float* conv2_w = (const float*)d_in[13];
    const float* conv2_b = (const float*)d_in[14];
    const float* out_w   = (const float*)d_in[15];
    const float* out_b   = (const float*)d_in[16];
    float* out = (float*)d_out;
    const int E = in_sizes[1];

    // workspace carve-up (floats)
    float* ws   = (float*)d_ws;
    float* deg  = ws;                                   // NT
    float* agg  = deg + NT;                             // NT*128
    float* feat = agg + (size_t)NT * F_IN;              // NT*97
    float* c1   = feat + (size_t)NT * TOTLAT;           // 64*16*30
    float* c2   = c1 + BGRAPH * 16 * KPOOL;             // 64*32*11
    int*   gidx = (int*)(c2 + BGRAPH * 32 * 11);        // 64*30

    const int TB = 256;
    const int GEMM_GRID = NT / (16 * 8);                // 750 blocks, 8 waves each

    // degrees
    init_deg_kernel<<<(NT + TB - 1) / TB, TB, 0, stream>>>(deg);
    deg_scatter_kernel<<<(E + TB - 1) / TB, TB, 0, stream>>>(edge_dst, deg, E);

    // ---- layer 1: K=128 from node_feat (aligned float4 fast path) ----
    agg_init_vec4_kernel<<<(NT * (F_IN / 4) + TB - 1) / TB, TB, 0, stream>>>(
        (float4*)agg, (const float4*)node_feat);
    {
        long long ts = (long long)E * (F_IN / 4);
        agg_scatter_vec4_kernel<<<(unsigned)((ts + TB - 1) / TB), TB, 0, stream>>>(
            agg, node_feat, edge_src, edge_dst, E);
    }
    gemm_tanh_wmma_kernel<128, 32><<<GEMM_GRID, TB, 0, stream>>>(
        agg, Wl[0], bl[0], deg, feat, 0);

    // ---- layers 2..4: K=32 from feat column slices ----
    const int cOff[4] = {0, 32, 64, 96};
    for (int L = 1; L < 4; ++L) {
        const int off = cOff[L - 1];
        long long ti = (long long)NT * 32;
        agg_init_kernel<<<(unsigned)((ti + TB - 1) / TB), TB, 0, stream>>>(
            agg, feat, TOTLAT, off, 32);
        long long ts = (long long)E * 32;
        agg_scatter_kernel<<<(unsigned)((ts + TB - 1) / TB), TB, 0, stream>>>(
            agg, feat, edge_src, edge_dst, TOTLAT, off, 32, E);
        if (L < 3)
            gemm_tanh_wmma_kernel<32, 32><<<GEMM_GRID, TB, 0, stream>>>(
                agg, Wl[L], bl[L], deg, feat, cOff[L]);
        else
            gemm_tanh_wmma_kernel<32, 1><<<GEMM_GRID, TB, 0, stream>>>(
                agg, Wl[L], bl[L], deg, feat, cOff[L]);
    }

    // sortpool + head
    sortpool_kernel<<<BGRAPH, TB, 0, stream>>>(feat, gidx);
    conv1_kernel<<<(BGRAPH * 16 * KPOOL + TB - 1) / TB, TB, 0, stream>>>(
        feat, gidx, conv1_w, conv1_b, c1);
    conv2_kernel<<<(BGRAPH * 32 * 11 + TB - 1) / TB, TB, 0, stream>>>(
        c1, conv2_w, conv2_b, c2);
    dense_kernel<<<(BGRAPH * OUTD + TB - 1) / TB, TB, 0, stream>>>(
        c2, out_w, out_b, out);
}
